// ResidualVQ_88021059764279
// MI455X (gfx1250) — matrix operand order repaired and checked by reading, not compiled, not run on hardware
//
#include <hip/hip_runtime.h>

// ---------------------------------------------------------------------------
// Residual VQ forward for MI455X (gfx1250, wave32, WMMA).
//   x:          [16, 2048, 256] f32  -> N = 32768 tokens, D = 256
//   embeddings: [8, 1024, 256]  f32  -> L = 8 layers, K = 1024 codes
// Outputs (flat in d_out, f32): quantized [N*D], loss [1], codes [N*L]
//
// Per-layer score GEMM via v_wmma_f32_16x16x32_bf16 with a bf16 hi/lo split
// (3 products -> fp32-class argmin accuracy at bf16 matrix throughput).
// Three independent accumulators (one per product term) break the WMMA RAW
// chain (depth 8 instead of 24 per 16x16 tile). Residual lives in the
// quantized output slot (quantized = x - final_residual).
// ---------------------------------------------------------------------------

typedef __attribute__((ext_vector_type(16))) __bf16 v16bf;
typedef __attribute__((ext_vector_type(8)))  float  v8f;

#define N_TOK   32768
#define DIM     256
#define KCODES  1024
#define LAYERS  8
#define ROWS_PER_BLOCK 128   // 8 waves * 16 rows
#define NTILES  (KCODES / 16)

static __device__ __forceinline__ __bf16 f2bf(float f) {
  // round-to-nearest-even f32 -> bf16 (values here are normal floats)
  unsigned u = __builtin_bit_cast(unsigned, f);
  unsigned r = (u + 0x7FFFu + ((u >> 16) & 1u)) >> 16;
  unsigned short s = (unsigned short)r;
  return __builtin_bit_cast(__bf16, s);
}
static __device__ __forceinline__ float bf2f(__bf16 h) {
  unsigned short s = __builtin_bit_cast(unsigned short, h);
  unsigned u = ((unsigned)s) << 16;
  return __builtin_bit_cast(float, u);
}

static __device__ __forceinline__ v8f wmma_bf16(v16bf a, v16bf b, v8f c) {
  return __builtin_amdgcn_wmma_f32_16x16x32_bf16(
      /*neg_a=*/false, a, /*neg_b=*/false, b,
      /*c_mod=*/(short)0, c, /*reuse_a=*/false, /*reuse_b=*/false);
}

union BPack { v16bf v; uint4 u[2]; };

// ---------------------------------------------------------------------------
// Prologue kernels
// ---------------------------------------------------------------------------

// R = x (float4 copy); also zero the loss accumulator.
__global__ void rvq_init_kernel(const float* __restrict__ x,
                                float* __restrict__ R,
                                float* __restrict__ loss, int n4) {
  int i = blockIdx.x * blockDim.x + threadIdx.x;
  if (i == 0) *loss = 0.0f;
  if (i < n4) ((float4*)R)[i] = ((const float4*)x)[i];
}

// enorm[code] = sum_d E[code][d]^2  (one wave per code)
__global__ __launch_bounds__(256) void rvq_enorm_kernel(
    const float* __restrict__ E, float* __restrict__ enorm) {
  int lane = threadIdx.x & 31;
  int code = (blockIdx.x * blockDim.x + threadIdx.x) >> 5;
  const float* row = E + (size_t)code * DIM;
  float s = 0.0f;
  #pragma unroll
  for (int i = 0; i < 8; ++i) { float v = row[lane + i * 32]; s += v * v; }
  #pragma unroll
  for (int m = 1; m < 32; m <<= 1) s += __shfl_xor(s, m, 32);
  if (lane == 0) enorm[code] = s;
}

// Split codebooks into bf16 hi/lo (fast path when ws is big enough)
__global__ void rvq_epack_kernel(const float* __restrict__ E,
                                 __bf16* __restrict__ Ehi,
                                 __bf16* __restrict__ Elo, int n) {
  int i = blockIdx.x * blockDim.x + threadIdx.x;
  if (i < n) {
    float f = E[i];
    __bf16 h = f2bf(f);
    Ehi[i] = h;
    Elo[i] = f2bf(f - bf2f(h));
  }
}

// quantized = x - final_residual (in place over R region of d_out)
__global__ void rvq_final_kernel(const float* __restrict__ x,
                                 float* __restrict__ Q, int n4) {
  int i = blockIdx.x * blockDim.x + threadIdx.x;
  if (i < n4) {
    float4 xv = ((const float4*)x)[i];
    float4 rv = ((float4*)Q)[i];
    float4 o;
    o.x = xv.x - rv.x; o.y = xv.y - rv.y; o.z = xv.z - rv.z; o.w = xv.w - rv.w;
    ((float4*)Q)[i] = o;
  }
}

// ---------------------------------------------------------------------------
// Fused per-layer kernel: WMMA score GEMM + argmin + gather + residual update
// One wave owns 16 token rows. Block = 8 waves = 128 rows. Grid = 256 blocks.
// Templated on PREPACK so the inner loop has no branch re-evaluation.
// ---------------------------------------------------------------------------
template <bool PREPACK>
__global__ __launch_bounds__(256) void rvq_layer_kernel(
    const float* __restrict__ E,       // this layer's codebook [K][D] f32
    const __bf16* __restrict__ Ehi,    // prepacked hi (PREPACK only)
    const __bf16* __restrict__ Elo,    // prepacked lo (PREPACK only)
    const float* __restrict__ enorm,   // [K]
    float* __restrict__ R,             // residual [N][D], updated in place
    float* __restrict__ codes,         // [N][L] (written as float values)
    float* __restrict__ loss,
    int layer) {
  const int lane    = threadIdx.x & 31;
  const int wave    = threadIdx.x >> 5;
  const int rowbase = blockIdx.x * ROWS_PER_BLOCK + wave * 16;
  const int mrow    = lane & 15;            // row / column-in-tile for this lane
  const int koff_a  = (lane >> 4) * 16;     // A layout: K groups of 16
  const int koff_b  = (lane >> 4) * 16;     // B layout: K groups of 16

  // ---- Load A tile (16 residual rows) and split into bf16 hi/lo ----
  // A 16x32 layout (matches B): lane<16 holds row M=lane, K = k0..k0+15
  // with k0=0; lane>=16 holds same row with K offset +16.
  v16bf a_hi[8], a_lo[8];
  const float* rrow = R + (size_t)(rowbase + mrow) * DIM;
  #pragma unroll
  for (int c = 0; c < 8; ++c) {
    const int k0 = c * 32 + koff_a;
    #pragma unroll
    for (int j = 0; j < 8; ++j) {
      float f0 = rrow[k0 + j];
      float f1 = rrow[k0 + 8 + j];
      __bf16 h0 = f2bf(f0), h1 = f2bf(f1);
      a_hi[c][j]     = h0;
      a_hi[c][j + 8] = h1;
      a_lo[c][j]     = f2bf(f0 - bf2f(h0));
      a_lo[c][j + 8] = f2bf(f1 - bf2f(h1));
    }
  }

  // ---- Scan all 64 code tiles, tracking per-lane best (score, index) ----
  float bestv[8];
  int   besti[8];
  #pragma unroll
  for (int j = 0; j < 8; ++j) { bestv[j] = 3.4e38f; besti[j] = 0; }

  for (int t = 0; t < NTILES; ++t) {
    const int brow = t * 16 + mrow;         // code row this lane supplies to B
    // Three independent accumulators: hi*hi, hi*lo, lo*hi product chains.
    v8f acc_hh = {0.f, 0.f, 0.f, 0.f, 0.f, 0.f, 0.f, 0.f};
    v8f acc_hl = {0.f, 0.f, 0.f, 0.f, 0.f, 0.f, 0.f, 0.f};
    v8f acc_lh = {0.f, 0.f, 0.f, 0.f, 0.f, 0.f, 0.f, 0.f};

    if (PREPACK) {                          // fast path: prepacked bf16 hi/lo
      const __bf16* eh = Ehi + (size_t)brow * DIM;
      const __bf16* el = Elo + (size_t)brow * DIM;
      #pragma unroll
      for (int c = 0; c < 8; ++c) {
        const int k0 = c * 32 + koff_b;
        BPack bh, bl;
        bh.u[0] = *(const uint4*)(eh + k0);
        bh.u[1] = *(const uint4*)(eh + k0 + 8);
        bl.u[0] = *(const uint4*)(el + k0);
        bl.u[1] = *(const uint4*)(el + k0 + 8);
        acc_hh = wmma_bf16(a_hi[c], bh.v, acc_hh);
        acc_hl = wmma_bf16(a_hi[c], bl.v, acc_hl);
        acc_lh = wmma_bf16(a_lo[c], bh.v, acc_lh);
      }
    } else {                                // fallback: split f32 on the fly
      const float* erow = E + (size_t)brow * DIM;
      #pragma unroll
      for (int c = 0; c < 8; ++c) {
        const int k0 = c * 32 + koff_b;
        v16bf bh, bl;
        #pragma unroll
        for (int j = 0; j < 16; ++j) {
          float f = erow[k0 + j];
          __bf16 h = f2bf(f);
          bh[j] = h;
          bl[j] = f2bf(f - bf2f(h));
        }
        acc_hh = wmma_bf16(a_hi[c], bh, acc_hh);
        acc_hl = wmma_bf16(a_hi[c], bl, acc_hl);
        acc_lh = wmma_bf16(a_lo[c], bh, acc_lh);
      }
    }

    // score = ||e||^2 - 2 * <r, e>   (||r||^2 constant over codes)
    const float en = enorm[brow];
    #pragma unroll
    for (int j = 0; j < 8; ++j) {
      float dot = acc_hh[j] + acc_hl[j] + acc_lh[j];
      float score = en - 2.0f * dot;
      if (score < bestv[j]) { bestv[j] = score; besti[j] = brow; }
    }
  }

  // ---- Cross-lane argmin within each 16-lane half (C layout: rows 0-7 in
  // lanes 0-15, rows 8-15 in lanes 16-31); ties -> smallest index ----
  #pragma unroll
  for (int j = 0; j < 8; ++j) {
    #pragma unroll
    for (int m = 1; m < 16; m <<= 1) {
      float ov = __shfl_xor(bestv[j], m, 32);
      int   oi = __shfl_xor(besti[j], m, 32);
      if (ov < bestv[j] || (ov == bestv[j] && oi < besti[j])) {
        bestv[j] = ov; besti[j] = oi;
      }
    }
  }

  // ---- Gather winning codes (exact f32), update residual, loss, codes ----
  float lsum = 0.0f;
  #pragma unroll
  for (int m = 0; m < 16; ++m) {
    const int src = (m < 8) ? 0 : 16;
    const int idx = __shfl(besti[m & 7], src, 32);   // broadcast to all lanes
    const float* e = E + (size_t)idx * DIM;
    float* rr = R + (size_t)(rowbase + m) * DIM;
    #pragma unroll
    for (int i = 0; i < 8; ++i) {
      const int d = lane + i * 32;
      float nr = rr[d] - e[d];     // new residual; also (r - q) for the loss
      rr[d] = nr;
      lsum += nr * nr;
    }
    if (lane == 0) codes[(size_t)(rowbase + m) * LAYERS + layer] = (float)idx;
  }

  #pragma unroll
  for (int m = 1; m < 32; m <<= 1) lsum += __shfl_xor(lsum, m, 32);
  if (lane == 0)
    atomicAdd(loss, lsum * (0.25f / (float)(N_TOK * DIM)));
}

// ---------------------------------------------------------------------------
extern "C" void kernel_launch(void* const* d_in, const int* in_sizes, int n_in,
                              void* d_out, int out_size, void* d_ws, size_t ws_size,
                              hipStream_t stream) {
  const float* x   = (const float*)d_in[0];
  const float* emb = (const float*)d_in[1];

  float* out   = (float*)d_out;
  float* R     = out;                 // residual lives in the quantized slot
  float* loss  = out + (size_t)N_TOK * DIM;
  float* codes = loss + 1;

  const size_t enorm_bytes = (size_t)LAYERS * KCODES * sizeof(float);
  const size_t pack_elems  = (size_t)LAYERS * KCODES * DIM;
  float*  enorm = (float*)d_ws;
  __bf16* ehi = nullptr;
  __bf16* elo = nullptr;
  const bool prepack =
      ws_size >= enorm_bytes + 2 * pack_elems * sizeof(__bf16);
  if (prepack) {
    ehi = (__bf16*)((char*)d_ws + enorm_bytes);
    elo = ehi + pack_elems;
  }

  const int n4 = N_TOK * DIM / 4;

  rvq_init_kernel<<<(n4 + 255) / 256, 256, 0, stream>>>(x, R, loss, n4);
  rvq_enorm_kernel<<<(LAYERS * KCODES * 32) / 256, 256, 0, stream>>>(emb, enorm);
  if (prepack)
    rvq_epack_kernel<<<(int)((pack_elems + 255) / 256), 256, 0, stream>>>(
        emb, ehi, elo, (int)pack_elems);

  for (int l = 0; l < LAYERS; ++l) {
    if (prepack) {
      rvq_layer_kernel<true><<<N_TOK / ROWS_PER_BLOCK, 256, 0, stream>>>(
          emb + (size_t)l * KCODES * DIM,
          ehi + (size_t)l * KCODES * DIM,
          elo + (size_t)l * KCODES * DIM,
          enorm + (size_t)l * KCODES, R, codes, loss, l);
    } else {
      rvq_layer_kernel<false><<<N_TOK / ROWS_PER_BLOCK, 256, 0, stream>>>(
          emb + (size_t)l * KCODES * DIM, nullptr, nullptr,
          enorm + (size_t)l * KCODES, R, codes, loss, l);
    }
  }

  rvq_final_kernel<<<(n4 + 255) / 256, 256, 0, stream>>>(x, R, n4);
}
